// ConcaveNN_35734127903346
// MI455X (gfx1250) — matrix operand order repaired and checked by reading, not compiled, not run on hardware
//
#include <hip/hip_runtime.h>
#include <math.h>

typedef _Float16 half_t;
typedef __attribute__((ext_vector_type(16))) _Float16 v16h;
typedef __attribute__((ext_vector_type(8)))  float    v8f;

#define NPT    51
#define NROWS  (NPT * NPT)            // 2601 rows per branch per sample
#define NTILES ((NROWS + 15) / 16)    // 163 row-tiles of 16
#define SPLIT  4

// workspace layout (float indices)
#define WS_CCW  0
#define WS_Q    64
#define WS_PART 128                    // SPLIT*128 partial sums
#define WS_COND (128 + SPLIT * 128)    // 128*2 : [2i]=offset, [2i+1]=scaling

// ---------------------------------------------------------------------------
// Reproduce the host-side float64 Clenshaw-Curtis weights/nodes exactly.
// ---------------------------------------------------------------------------
__global__ void cc_setup_kernel(float* ws) {
  const int j = threadIdx.x;
  const double PI = 3.14159265358979323846;
  if (j <= 50) {
    double s = 0.0;
    for (int i = 0; i <= 50; ++i) {
      double lam;
      if (j == 0) lam = 0.5;
      else        lam = cos((double)(i * j) * PI / 50.0);
      if (j == 50) lam *= 0.5;
      lam *= 2.0 / 50.0;
      double W;
      if (i == 0)      W = 1.0;
      else if (i & 1)  W = 0.0;
      else             W = 2.0 / (1.0 - (double)(i * i));
      s += lam * W;
    }
    ws[WS_CCW + j] = (float)s;
    ws[WS_Q + j]   = (float)((cos((double)j * PI / 50.0) + 1.0) * 0.5);
  }
}

// ---------------------------------------------------------------------------
// Main kernel: one (sample, split) pair per block; 128 threads = 4 waves.
// ---------------------------------------------------------------------------
__global__ __launch_bounds__(128, 1)
void umnn_main_kernel(const float* __restrict__ x,  const float* __restrict__ h,
                      const float* __restrict__ pw0, const float* __restrict__ pb0,
                      const float* __restrict__ pw1, const float* __restrict__ pb1,
                      const float* __restrict__ pw2, const float* __restrict__ pb2,
                      const float* __restrict__ nw0, const float* __restrict__ nb0,
                      const float* __restrict__ nw1, const float* __restrict__ nb1,
                      const float* __restrict__ nw2, const float* __restrict__ nb2,
                      const float* __restrict__ cw0, const float* __restrict__ cb0,
                      const float* __restrict__ cw1, const float* __restrict__ cb1,
                      const float* __restrict__ cw2, const float* __restrict__ cb2,
                      float* __restrict__ ws) {
  // W1 of the current branch, pre-swizzled into WMMA B-fragment order:
  // frag = kk*8+nt ; lane = ((k>>4)&1)*16 + (n&15) ; element = k&15
  __shared__ v16h Bswz[32][32];                       // 32 KB
  __shared__ __align__(16) float aw[2][4][2][16];     // w0u, A-frag swizzled
  __shared__ __align__(16) float ab[2][4][2][16];     // base0, A-frag swizzled
  __shared__ float b1s[2][128];
  __shared__ float w2s[2][128];
  __shared__ float b2s[2];
  __shared__ float ccw[NPT];
  __shared__ float qv[NPT];
  __shared__ float hh[32];
  __shared__ float zc[128];
  __shared__ float zc2[128];
  __shared__ float wred[4];

  const int tid    = threadIdx.x;
  const int lane   = tid & 31;
  const int wave   = tid >> 5;
  const int lanelo = lane & 15;
  const int hiIdx  = (lane >> 4) & 1;
  const int bid    = blockIdx.x;   // sample
  const int sblk   = blockIdx.y;   // split

  if (tid < 32)  hh[tid] = h[bid * 32 + tid];
  if (tid < NPT) { ccw[tid] = ws[WS_CCW + tid]; qv[tid] = ws[WS_Q + tid]; }
  b1s[0][tid] = pb1[tid];  b1s[1][tid] = nb1[tid];
  w2s[0][tid] = pw2[tid];  w2s[1][tid] = nw2[tid];
  if (tid == 0) { b2s[0] = pb2[0]; b2s[1] = nb2[0]; }

  // block-wide max(x) -> integration upper bound T = max(x) + 10
  {
    float xv = x[tid];
#pragma unroll
    for (int m = 1; m <= 16; m <<= 1) xv = fmaxf(xv, __shfl_xor(xv, m, 32));
    if (lane == 0) wred[wave] = xv;
  }
  __syncthreads();
  const float Tmax = fmaxf(fmaxf(wred[0], wred[1]), fmaxf(wred[2], wred[3])) + 10.0f;
  const float xi   = x[bid];

  // layer-0 rank-1 params: z0 = relu(u*w0u + base0), swizzled into A-frag order
  {
    const int c = tid;
    float s0 = pb0[c], s1 = nb0[c];
    for (int d = 0; d < 32; ++d) {
      s0 += hh[d] * pw0[(d + 1) * 128 + c];
      s1 += hh[d] * nw0[(d + 1) * 128 + c];
    }
    const int kk = c >> 5, r = c & 31;
    const int hi = (r >> 4) & 1;
    const int e  = r & 15;
    aw[0][kk][hi][e] = pw0[c];  ab[0][kk][hi][e] = s0;
    aw[1][kk][hi][e] = nw0[c];  ab[1][kk][hi][e] = s1;
  }

  // conditioner MLP (h -> offset, scaling), done once per sample (sblk==0)
  if (sblk == 0) {
    float a0 = cb0[tid];
    for (int d = 0; d < 32; ++d) a0 += hh[d] * cw0[d * 128 + tid];
    zc[tid] = fmaxf(a0, 0.0f);
    __syncthreads();
    float a1 = cb1[tid];
    for (int d = 0; d < 128; ++d) a1 += zc[d] * cw1[d * 128 + tid];
    zc2[tid] = fmaxf(a1, 0.0f);
    __syncthreads();
    if (tid < 2) {
      float o = cb2[tid];
      for (int d = 0; d < 128; ++d) o += zc2[d] * cw2[d * 2 + tid];
      ws[WS_COND + bid * 2 + tid] = (tid == 1) ? expf(o) : o;
    }
  }

  float Sacc = 0.0f;

  for (int br = 0; br < 2; ++br) {
    __syncthreads();   // previous-branch Bswz readers done
    const float* __restrict__ W1 = br ? nw1 : pw1;
    for (int idx = tid; idx < 128 * 128; idx += 128) {
      const int k = idx >> 7, n = idx & 127;
      const int frag  = ((k >> 5) << 3) + (n >> 4);
      const int lane2 = (((k >> 4) & 1) << 4) + (n & 15);
      ((half_t*)&Bswz[frag][lane2])[k & 15] = (half_t)W1[idx];
    }
    __syncthreads();

    // keep all 32 B fragments (full 128x128 W1) resident in registers
    v16h Bf[32];
#pragma unroll
    for (int f = 0; f < 32; ++f) Bf[f] = Bswz[f][lane];

    const float b2v  = b2s[br];
    const float sgn  = br ? -1.0f : 1.0f;
    const int   slot = sblk * 4 + wave;                 // 0..15
    const int   me   = lanelo + ((lane & 16) ? 8 : 0);  // epilogue row

    for (int tile = slot; tile < NTILES; tile += 4 * SPLIT) {
      // u (integrand abscissa) for this lane's A-matrix row
      float u = 0.0f;
      {
        const int r = tile * 16 + lanelo;
        if (r < NROWS) {
          const int j = r / NPT, k = r - j * NPT;
          const float t = xi * qv[j];
          u = br ? (t * qv[k]) : (t + (Tmax - t) * qv[k]);
        }
      }
      // quadrature weight for this lane's epilogue row
      float wgt = 0.0f;
      {
        const int re = tile * 16 + me;
        if (lanelo < 8 && re < NROWS) {
          const int j = re / NPT, k = re - j * NPT;
          const float t    = xi * qv[j];
          const float span = br ? t : (Tmax - t);
          wgt = ccw[k] * ccw[j] * span * 0.25f * xi;
        }
      }

      v8f acc[8];
#pragma unroll
      for (int nt = 0; nt < 8; ++nt) {
#pragma unroll
        for (int q = 0; q < 8; ++q) acc[nt][q] = 0.0f;
      }

#pragma unroll
      for (int kk = 0; kk < 4; ++kk) {
        // build A fragment: z0 = relu(u*w0u + base0) for 32 K-channels
        v16h a;
        const float4* awp = (const float4*)&aw[br][kk][hiIdx][0];
        const float4* abp = (const float4*)&ab[br][kk][hiIdx][0];
#pragma unroll
        for (int q = 0; q < 4; ++q) {
          const float4 wv = awp[q];
          const float4 bv = abp[q];
          float z;
          z = fmaf(u, wv.x, bv.x); a[4 * q + 0] = (half_t)fmaxf(z, 0.0f);
          z = fmaf(u, wv.y, bv.y); a[4 * q + 1] = (half_t)fmaxf(z, 0.0f);
          z = fmaf(u, wv.z, bv.z); a[4 * q + 2] = (half_t)fmaxf(z, 0.0f);
          z = fmaf(u, wv.w, bv.w); a[4 * q + 3] = (half_t)fmaxf(z, 0.0f);
        }
#pragma unroll
        for (int nt = 0; nt < 8; ++nt) {
          acc[nt] = __builtin_amdgcn_wmma_f32_16x16x32_f16(
              false, a, false, Bf[kk * 8 + nt], (short)0, acc[nt], false, false);
        }
      }

      // epilogue: relu(z1 + b1) . w2, reduced over the 128 columns
      float p[8];
#pragma unroll
      for (int rr = 0; rr < 8; ++rr) p[rr] = 0.0f;
#pragma unroll
      for (int nt = 0; nt < 8; ++nt) {
        const int   c  = nt * 16 + lanelo;
        const float bb = b1s[br][c];
        const float ww = w2s[br][c];
#pragma unroll
        for (int rr = 0; rr < 8; ++rr)
          p[rr] += fmaxf(acc[nt][rr] + bb, 0.0f) * ww;
      }
#pragma unroll
      for (int m = 1; m <= 8; m <<= 1) {
#pragma unroll
        for (int rr = 0; rr < 8; ++rr) p[rr] += __shfl_xor(p[rr], m, 32);
      }
      float sfin = p[0];
#pragma unroll
      for (int rr = 1; rr < 8; ++rr)
        if ((lanelo & 7) == rr) sfin = p[rr];

      const float z2  = sfin + b2v;
      const float val = (z2 > 0.0f) ? z2 : (expf(z2) - 1.0f);  // elu
      Sacc += sgn * (val + 1.0f) * wgt;   // pos: (elu+1)*w ; neg: -(elu+1)*w
    }
  }

  // block reduction of per-lane quadrature sums -> deterministic partial
#pragma unroll
  for (int m = 1; m <= 16; m <<= 1) Sacc += __shfl_xor(Sacc, m, 32);
  __syncthreads();
  if (lane == 0) wred[wave] = Sacc;
  __syncthreads();
  if (tid == 0)
    ws[WS_PART + sblk * 128 + bid] = wred[0] + wred[1] + wred[2] + wred[3];
}

// ---------------------------------------------------------------------------
// Combine split partials with conditioner offset/scaling (deterministic).
// ---------------------------------------------------------------------------
__global__ void umnn_finish_kernel(const float* __restrict__ ws,
                                   float* __restrict__ out) {
  const int i = threadIdx.x;
  float s = 0.0f;
  for (int p = 0; p < SPLIT; ++p) s += ws[WS_PART + p * 128 + i];
  out[i] = ws[WS_COND + 2 * i + 1] * s + ws[WS_COND + 2 * i];
}

extern "C" void kernel_launch(void* const* d_in, const int* in_sizes, int n_in,
                              void* d_out, int out_size, void* d_ws, size_t ws_size,
                              hipStream_t stream) {
  (void)in_sizes; (void)n_in; (void)out_size; (void)ws_size;
  const float* x   = (const float*)d_in[0];
  const float* h   = (const float*)d_in[1];
  const float* pw0 = (const float*)d_in[2];
  const float* pb0 = (const float*)d_in[3];
  const float* pw1 = (const float*)d_in[4];
  const float* pb1 = (const float*)d_in[5];
  const float* pw2 = (const float*)d_in[6];
  const float* pb2 = (const float*)d_in[7];
  const float* nw0 = (const float*)d_in[8];
  const float* nb0 = (const float*)d_in[9];
  const float* nw1 = (const float*)d_in[10];
  const float* nb1 = (const float*)d_in[11];
  const float* nw2 = (const float*)d_in[12];
  const float* nb2 = (const float*)d_in[13];
  const float* cw0 = (const float*)d_in[14];
  const float* cb0 = (const float*)d_in[15];
  const float* cw1 = (const float*)d_in[16];
  const float* cb1 = (const float*)d_in[17];
  const float* cw2 = (const float*)d_in[18];
  const float* cb2 = (const float*)d_in[19];
  float* ws = (float*)d_ws;

  cc_setup_kernel<<<1, 64, 0, stream>>>(ws);
  umnn_main_kernel<<<dim3(128, SPLIT), 128, 0, stream>>>(
      x, h, pw0, pb0, pw1, pb1, pw2, pb2,
      nw0, nb0, nw1, nb1, nw2, nb2,
      cw0, cb0, cw1, cb1, cw2, cb2, ws);
  umnn_finish_kernel<<<1, 128, 0, stream>>>(ws, (float*)d_out);
}
